// GraphMSG_57011395887381
// MI455X (gfx1250) — matrix operand order repaired and checked by reading, not compiled, not run on hardware
//
#include <hip/hip_runtime.h>
#include <hip/hip_bf16.h>

// ===========================================================================
// GraphCast-style GNN forward for MI455X (gfx1250, wave32, WMMA).
//
// GEMMs: v_wmma_f32_16x16x32_bf16; A tile staged in LDS (async-to-LDS copy
// on the bf16 path), B fragments loaded straight from global (W is hot in
// WGP$/L2; removes all barriers from the K loop). Edge/node MLP layer-1
// fuses the index-gather + concat (f32 -> bf16) into the LDS A-tile.
// Segment-sum via global_atomic_add_f32.
//
// Workspace requirement: ~345 MB. Param flattening assumption:
// setup_inputs() dict order, params flattened depth-first in insertion
// order (W0,W1,b0,b1,ln_g,ln_b per MLP).
// ===========================================================================

constexpr int MT     = 64;    // M rows per workgroup tile
constexpr int TPB    = 128;   // 4 waves of 32
constexpr int EPITCH = 132;   // f32 epilogue LDS row pitch (128 + 4 pad)

constexpr int cN_ERA = 40320;
constexpr int cN_H   = 10242;
constexpr int cE_E2H = 196608;
constexpr int cE_H2E = 122880;
constexpr int cE_H2H = 81920;

typedef __bf16 bf16;
typedef __attribute__((ext_vector_type(16))) __bf16 v16bf;
typedef __attribute__((ext_vector_type(8)))  float  v8f;

union FragBF { uint4 u[2]; v16bf v; };
union Pack8  { bf16 h[8]; uint4 u; };
union Pack4  { bf16 h[4]; uint2 u; };

static __device__ __forceinline__ bf16  tobf(float f)  { return (bf16)f; }
static __device__ __forceinline__ float siluf(float v) { return v / (1.f + __expf(-v)); }

// CDNA5 async global->LDS copy (16B per lane), tracked by ASYNCcnt.
static __device__ __forceinline__ void async_copy16(unsigned lds_off, const void* gaddr) {
  asm volatile("global_load_async_to_lds_b128 %0, %1, off"
               :: "v"(lds_off), "v"((unsigned long long)(uintptr_t)gaddr)
               : "memory");
}
static __device__ __forceinline__ void wait_async0() {
  asm volatile("s_wait_asynccnt 0" ::: "memory");
}

// ---------------------------------------------------------------------------
// WMMA main loop. A tile (MT x Kpad bf16) in LDS; B fragments straight from
// global W[Kpad,Npad] (row-major bf16). Each wave: 16 x (NF*16) tile.
//
// A fragment (16-bit A 16x32, ISA 7.12.2): lane L -> row L&15;
//   halves 0..7 = K kb+koff..+7, halves 8..15 = K kb+koff+16..+23,
//   koff = (L>=16) ? 8 : 0.
// B fragment (32x16): lane L -> col N = n0+(L&15), halves same K order as A.
// ---------------------------------------------------------------------------
template<int NF>
static __device__ __forceinline__ void gemm_core(
    const bf16* __restrict__ Wg, int Kpad, int Npad,
    const bf16* At, int aPitch, v8f* acc)
{
  const int tid  = threadIdx.x;
  const int lane = tid & 31;
  const int wid  = tid >> 5;
  const bf16* arow = At + ((wid << 4) + (lane & 15)) * aPitch + ((lane & 16) ? 8 : 0);
  const bf16* wrow = Wg + (lane & 15) + (size_t)((lane & 16) ? 8 : 0) * Npad;
  __syncthreads();                       // A tile staged
  FragBF a;
  a.u[0] = *(const uint4*)(arow);
  a.u[1] = *(const uint4*)(arow + 16);
  for (int kb = 0; kb < Kpad; kb += 32) {
    FragBF b[NF];
    const bf16* wk = wrow + (size_t)kb * Npad;
#pragma unroll
    for (int nf = 0; nf < NF; ++nf) {
#pragma unroll
      for (int s = 0; s < 16; ++s)
        b[nf].v[s] = wk[(nf << 4) + (size_t)(s + (s & 8)) * Npad];
    }
    FragBF an = a;                       // software-pipelined A fragment
    if (kb + 32 < Kpad) {
      an.u[0] = *(const uint4*)(arow + kb + 32);
      an.u[1] = *(const uint4*)(arow + kb + 48);
    }
#pragma unroll
    for (int nf = 0; nf < NF; ++nf)
      acc[nf] = __builtin_amdgcn_wmma_f32_16x16x32_bf16(
          false, a.v, false, b[nf].v, (short)0, acc[nf], false, false);
    a = an;
  }
}

// ---------------------------------------------------------------------------
// Epilogue: accumulators -> LDS (C layout: VGPR j, lane L -> row j +
// (L>=16?8:0), col L&15), then one thread per row: bias -> SiLU ->
// LayerNorm -> residual; float4-vectorized.
// ---------------------------------------------------------------------------
template<int NF>
static __device__ __forceinline__ void epilogue(
    v8f* acc, float* eL, int row0, int M, int N, int Npad,
    const float* __restrict__ bias, int act,
    const float* __restrict__ ln_g, const float* __restrict__ ln_b,
    const float* __restrict__ res, int res_ld,
    float* __restrict__ outF, int ldoF,
    bf16* __restrict__ outH, int ldoH)
{
  const int tid = threadIdx.x, lane = tid & 31, wid = tid >> 5;
  const int rb = (wid << 4) + ((lane & 16) ? 8 : 0);
  const int c0 = lane & 15;
  __syncthreads();                       // all WMMA / LDS reads done
#pragma unroll
  for (int nf = 0; nf < NF; ++nf)
#pragma unroll
    for (int j = 0; j < 8; ++j)
      eL[(rb + j) * EPITCH + (nf << 4) + c0] = acc[nf][j];
  __syncthreads();
  if (tid < MT) {
    const int grow = row0 + tid;
    if (grow < M) {
      float* rp = eL + tid * EPITCH;
      for (int c = 0; c < Npad; c += 4) {
        float4 v = *(float4*)(rp + c);
        float4 b4 = *(const float4*)(bias + c);
        v.x += b4.x; v.y += b4.y; v.z += b4.z; v.w += b4.w;
        if (act) { v.x = siluf(v.x); v.y = siluf(v.y); v.z = siluf(v.z); v.w = siluf(v.w); }
        *(float4*)(rp + c) = v;
      }
      if (ln_g) {
        float mu = 0.f;
        for (int c = 0; c < N; c += 4) {
          float4 v = *(float4*)(rp + c); mu += v.x + v.y + v.z + v.w;
        }
        mu *= (1.f / N);
        float s2 = 0.f;
        for (int c = 0; c < N; c += 4) {
          float4 v = *(float4*)(rp + c);
          float dx = v.x - mu, dy = v.y - mu, dz = v.z - mu, dw = v.w - mu;
          s2 += dx * dx + dy * dy + dz * dz + dw * dw;
        }
        float rstd = __frsqrt_rn(s2 * (1.f / N) + 1e-5f);
        for (int c = 0; c < N; c += 4) {
          float4 v = *(float4*)(rp + c);
          float4 g4 = *(const float4*)(ln_g + c);
          float4 o4 = *(const float4*)(ln_b + c);
          v.x = (v.x - mu) * rstd * g4.x + o4.x;
          v.y = (v.y - mu) * rstd * g4.y + o4.y;
          v.z = (v.z - mu) * rstd * g4.z + o4.z;
          v.w = (v.w - mu) * rstd * g4.w + o4.w;
          *(float4*)(rp + c) = v;
        }
      }
      for (int c = 0; c < N; c += 4) {
        float4 v = *(float4*)(rp + c);
        if (res) {
          float4 r4 = *(const float4*)(res + (size_t)grow * res_ld + c);
          v.x += r4.x; v.y += r4.y; v.z += r4.z; v.w += r4.w;
        }
        if (outF) *(float4*)(outF + (size_t)grow * ldoF + c) = v;
        if (outH) {
          Pack4 p;
          p.h[0] = tobf(v.x); p.h[1] = tobf(v.y); p.h[2] = tobf(v.z); p.h[3] = tobf(v.w);
          *(uint2*)(outH + (size_t)grow * ldoH + c) = p.u;
        }
      }
    }
  }
}

// ---------------------------------------------------------------------------
// Generic GEMM: Y = post( A[M,K] @ W[Kpad,Npad] + b ). bf16 A staged via
// async-to-LDS; f32 A converted while staging.
// NOTE: (MT*Kpad/8) is an exact multiple of 2*TPB for every Kpad used.
// ---------------------------------------------------------------------------
template<int NF>
__global__ void __launch_bounds__(TPB) gemm_kernel(
    const void* __restrict__ A, int a_is_f32, int lda,
    const bf16* __restrict__ Wg, int Kpad, int Npad, int N,
    const float* __restrict__ bias, int act,
    const float* __restrict__ ln_g, const float* __restrict__ ln_b,
    const float* __restrict__ res, int res_ld,
    float* __restrict__ outF, int ldoF,
    bf16* __restrict__ outH, int ldoH, int M)
{
  extern __shared__ char smem[];
  const int aPitch = Kpad + 8;
  bf16* At = (bf16*)smem;
  const int tid  = threadIdx.x;
  const int row0 = blockIdx.x * MT;
  const int chunks = (MT * Kpad) >> 3;

  // warm WGP$/L2 with the weight matrix
  {
    const char* wp = (const char*)Wg;
    const size_t wbytes = (size_t)Kpad * Npad * sizeof(bf16);
    for (size_t o = (size_t)tid * 64; o < wbytes; o += (size_t)TPB * 64)
      __builtin_prefetch(wp + o, 0, 0);
  }

  if (a_is_f32) {
    const float* Af = (const float*)A;
    auto stage = [&](int c) {
      int h = c << 3; int r = h / Kpad; int k = h - r * Kpad;
      Pack8 p; int grow = row0 + r;
      if (grow < M) {
        const float4* s4 = (const float4*)(Af + (size_t)grow * lda + k);
        float4 f0 = s4[0], f1 = s4[1];
        p.h[0] = tobf(f0.x); p.h[1] = tobf(f0.y); p.h[2] = tobf(f0.z); p.h[3] = tobf(f0.w);
        p.h[4] = tobf(f1.x); p.h[5] = tobf(f1.y); p.h[6] = tobf(f1.z); p.h[7] = tobf(f1.w);
      } else { p.u.x = p.u.y = p.u.z = p.u.w = 0; }
      *(uint4*)(At + r * aPitch + k) = p.u;
    };
    for (int c = tid; c < chunks; c += 2 * TPB) { stage(c); stage(c + TPB); }
  } else {
    const bf16* Ah = (const bf16*)A;
    for (int c = tid; c < chunks; c += TPB) {
      int h = c << 3; int r = h / Kpad; int k = h - r * Kpad;
      int grow = row0 + r;
      if (grow >= M) grow = M - 1;   // clamp: OOB rows feed only masked outputs
      async_copy16((unsigned)(uintptr_t)(At + r * aPitch + k),
                   Ah + (size_t)grow * lda + k);
    }
    wait_async0();
  }

  v8f acc[NF];
#pragma unroll
  for (int i = 0; i < NF; ++i)
#pragma unroll
    for (int j = 0; j < 8; ++j) acc[i][j] = 0.f;
  gemm_core<NF>(Wg, Kpad, Npad, At, aPitch, acc);
  epilogue<NF>(acc, (float*)smem, row0, M, N, Npad, bias, act,
               ln_g, ln_b, res, res_ld, outF, ldoF, outH, ldoH);
}

// ---------------------------------------------------------------------------
// Fused edge-MLP layer 1: A-row = [ x_dst[ei1[e]] | x_src[ei0[e]] | ea[e] ]
// (K=384, f32 gathered -> bf16 LDS), out = silu(A@W0+b0) as bf16 hidden.
// ---------------------------------------------------------------------------
__global__ void __launch_bounds__(TPB) edge_l1_kernel(
    const int* __restrict__ ei, int E,
    const float* __restrict__ xsrc, const float* __restrict__ xdst,
    const float* __restrict__ ea,
    const bf16* __restrict__ Wg, const float* __restrict__ bias,
    bf16* __restrict__ outH)
{
  constexpr int KP = 384;
  extern __shared__ char smem[];
  const int aPitch = KP + 8;
  bf16* At = (bf16*)smem;
  const int tid  = threadIdx.x;
  const int row0 = blockIdx.x * MT;
  auto stage = [&](int c) {
    int h = c << 3; int r = h / KP; int k = h - r * KP;
    Pack8 p; int e = row0 + r;
    if (e < E) {
      int seg = k >> 7, kl = k & 127;
      const float* src;
      if (seg == 0)      src = xdst + (size_t)ei[E + e] * 128 + kl;  // x_i
      else if (seg == 1) src = xsrc + (size_t)ei[e]     * 128 + kl;  // x_j
      else               src = ea   + (size_t)e         * 128 + kl;  // ea
      const float4* s4 = (const float4*)src;
      float4 f0 = s4[0], f1 = s4[1];
      p.h[0] = tobf(f0.x); p.h[1] = tobf(f0.y); p.h[2] = tobf(f0.z); p.h[3] = tobf(f0.w);
      p.h[4] = tobf(f1.x); p.h[5] = tobf(f1.y); p.h[6] = tobf(f1.z); p.h[7] = tobf(f1.w);
    } else { p.u.x = p.u.y = p.u.z = p.u.w = 0; }
    *(uint4*)(At + r * aPitch + k) = p.u;
  };
  for (int c = tid; c < (MT * KP) >> 3; c += 2 * TPB) { stage(c); stage(c + TPB); }
  v8f acc[8];
#pragma unroll
  for (int i = 0; i < 8; ++i)
#pragma unroll
    for (int j = 0; j < 8; ++j) acc[i][j] = 0.f;
  gemm_core<8>(Wg, KP, 128, At, aPitch, acc);
  epilogue<8>(acc, (float*)smem, row0, E, 128, 128, bias, 1,
              nullptr, nullptr, nullptr, 0, nullptr, 0, outH, 128);
}

// ---------------------------------------------------------------------------
// Fused node-MLP layer 1: A-row = [ x_dst[r] | agg[r] ] (K=256).
// ---------------------------------------------------------------------------
__global__ void __launch_bounds__(TPB) node_l1_kernel(
    const float* __restrict__ xdst, const float* __restrict__ agg,
    const bf16* __restrict__ Wg, const float* __restrict__ bias,
    bf16* __restrict__ outH, int M)
{
  constexpr int KP = 256;
  extern __shared__ char smem[];
  const int aPitch = KP + 8;
  bf16* At = (bf16*)smem;
  const int tid  = threadIdx.x;
  const int row0 = blockIdx.x * MT;
  auto stage = [&](int c) {
    int h = c << 3; int r = h / KP; int k = h - r * KP;
    Pack8 p; int grow = row0 + r;
    if (grow < M) {
      int seg = k >> 7, kl = k & 127;
      const float* src = (seg == 0) ? xdst + (size_t)grow * 128 + kl
                                    : agg  + (size_t)grow * 128 + kl;
      const float4* s4 = (const float4*)src;
      float4 f0 = s4[0], f1 = s4[1];
      p.h[0] = tobf(f0.x); p.h[1] = tobf(f0.y); p.h[2] = tobf(f0.z); p.h[3] = tobf(f0.w);
      p.h[4] = tobf(f1.x); p.h[5] = tobf(f1.y); p.h[6] = tobf(f1.z); p.h[7] = tobf(f1.w);
    } else { p.u.x = p.u.y = p.u.z = p.u.w = 0; }
    *(uint4*)(At + r * aPitch + k) = p.u;
  };
  for (int c = tid; c < (MT * KP) >> 3; c += 2 * TPB) { stage(c); stage(c + TPB); }
  v8f acc[8];
#pragma unroll
  for (int i = 0; i < 8; ++i)
#pragma unroll
    for (int j = 0; j < 8; ++j) acc[i][j] = 0.f;
  gemm_core<8>(Wg, KP, 128, At, aPitch, acc);
  epilogue<8>(acc, (float*)smem, row0, M, 128, 128, bias, 1,
              nullptr, nullptr, nullptr, 0, nullptr, 0, outH, 128);
}

// ------------------------- small helper kernels ----------------------------
__global__ void scatter_add_kernel(const float* __restrict__ ea,
                                   const int* __restrict__ ei, int E,
                                   float* __restrict__ agg)
{
  size_t t = (size_t)blockIdx.x * blockDim.x + threadIdx.x;
  if (t >= (size_t)E * 128) return;
  int e = (int)(t >> 7), c = (int)(t & 127);
  atomicAdd(agg + (size_t)ei[E + e] * 128 + c, ea[t]);
}

__global__ void zero_f32_kernel(float* p, int n)
{ int t = blockIdx.x * blockDim.x + threadIdx.x; if (t < n) p[t] = 0.f; }

__global__ void copy_f32_kernel(float* d, const float* s, int n)
{ int t = blockIdx.x * blockDim.x + threadIdx.x; if (t < n) d[t] = s[t]; }

__global__ void addto_f32_kernel(float* d, const float* s, int n)
{ int t = blockIdx.x * blockDim.x + threadIdx.x; if (t < n) d[t] += s[t]; }

// pad + convert weights f32[Ks,Ns] -> bf16[Kp,Np] (zero padded)
__global__ void pad_w_kernel(const float* __restrict__ src, int Ks, int Ns,
                             bf16* __restrict__ dst, int Kp, int Np)
{
  int t = blockIdx.x * blockDim.x + threadIdx.x;
  if (t >= Kp * Np) return;
  int k = t / Np, n = t - k * Np;
  dst[t] = (k < Ks && n < Ns) ? tobf(src[k * Ns + n]) : tobf(0.f);
}

__global__ void pad_b_kernel(const float* __restrict__ src, int Ns,
                             float* __restrict__ dst, int Np)
{
  int t = blockIdx.x * blockDim.x + threadIdx.x;
  if (t < Np) dst[t] = (t < Ns) ? src[t] : 0.f;
}

// x_era_in = [ x('b m n f -> (b n) (m f)') | era_ll | era_tr | 0-pad ], K=192
__global__ void prep_era_in_kernel(const float* __restrict__ x,
                                   const float* __restrict__ ll,
                                   const float* __restrict__ tr,
                                   bf16* __restrict__ out)
{
  int t = blockIdx.x * blockDim.x + threadIdx.x;
  if (t >= cN_ERA * 192) return;
  int n = t / 192, k = t - n * 192;
  float v = 0.f;
  if (k < 160) { int m = k / 80, f = k - m * 80;
                 v = x[((size_t)m * cN_ERA + n) * 80 + f]; }
  else if (k < 164) v = ll[n * 4 + (k - 160)];
  else if (k < 172) v = tr[n * 8 + (k - 164)];
  out[t] = tobf(v);
}

// x_h_in = [ h_ll | h_tr | 0-pad ], K=32
__global__ void prep_h_in_kernel(const float* __restrict__ ll,
                                 const float* __restrict__ tr,
                                 bf16* __restrict__ out)
{
  int t = blockIdx.x * blockDim.x + threadIdx.x;
  if (t >= cN_H * 32) return;
  int n = t >> 5, k = t & 31;
  float v = 0.f;
  if (k < 4)       v = ll[n * 4 + k];
  else if (k < 12) v = tr[n * 8 + (k - 4)];
  out[t] = tobf(v);
}

// ea_in = [ edge_attr(3) | edge_tr(8) | 0-pad ], K=32
__global__ void prep_edge_in_kernel(const float* __restrict__ attr,
                                    const float* __restrict__ tr,
                                    bf16* __restrict__ out, int E)
{
  int t = blockIdx.x * blockDim.x + threadIdx.x;
  if (t >= E * 32) return;
  int e = t >> 5, k = t & 31;
  float v = 0.f;
  if (k < 3)       v = attr[e * 3 + k];
  else if (k < 11) v = tr[e * 8 + (k - 3)];
  out[t] = tobf(v);
}

// =========================== host orchestration ============================
struct MLP2 {
  const bf16 *W0, *W1;
  const float *b0, *b1, *g, *bln;
  int K0p;
};

extern "C" void kernel_launch(void* const* d_in, const int* in_sizes, int n_in,
                              void* d_out, int out_size, void* d_ws, size_t ws_size,
                              hipStream_t stream)
{
  (void)in_sizes; (void)out_size; (void)ws_size;
  if (n_in < 120) return;   // param-flattening assumption violated

  auto F = [&](int i) { return (const float*)d_in[i]; };
  auto I = [&](int i) { return (const int*)d_in[i]; };

  const float* x       = F(0);
  const int*   e2h_ei  = I(1);
  const int*   h2e_ei  = I(2);
  const int*   h2h_ei  = I(3);
  const float* e2h_at  = F(4);
  const float* h2e_at  = F(5);
  const float* h2h_at  = F(6);
  const float* era_ll  = F(7);
  const float* h_ll    = F(8);

  // ---- workspace bump allocator (256B aligned) ----
  char* wsp = (char*)d_ws;
  auto alloc = [&](size_t bytes) -> char* {
    char* p = wsp; wsp += (bytes + 255) & ~(size_t)255; return p;
  };

  bf16* wpool = (bf16*)alloc((size_t)1200000 * sizeof(bf16));
  bf16* wcur  = wpool;
  auto cvtW = [&](const float* W, int Ks, int Ns, int Kp, int Np) -> const bf16* {
    bf16* dst = wcur; wcur += (size_t)Kp * Np;
    int tot = Kp * Np;
    pad_w_kernel<<<(tot + 255) / 256, 256, 0, stream>>>(W, Ks, Ns, dst, Kp, Np);
    return dst;
  };

  // ---- params (flattened depth-first, insertion order) ----
  int ix = 9;
  const float* era_tr = F(ix++);
  const float* h_tr   = F(ix++);
  const float* e2h_tr = F(ix++);
  const float* h2e_tr = F(ix++);
  const float* h2h_tr = F(ix++);

  auto readMLP = [&](int Ks0, int Kp0) -> MLP2 {
    MLP2 m;
    const float* W0 = F(ix++); const float* W1 = F(ix++);
    m.b0 = F(ix++); m.b1 = F(ix++); m.g = F(ix++); m.bln = F(ix++);
    m.W0 = cvtW(W0, Ks0, 128, Kp0, 128);
    m.W1 = cvtW(W1, 128, 128, 128, 128);
    m.K0p = Kp0;
    return m;
  };

  MLP2 f_src   = readMLP(172, 192);
  MLP2 f_dst   = readMLP(12, 32);
  MLP2 f_edges = readMLP(11, 32);
  MLP2 f_bedge = readMLP(384, 384);
  MLP2 f_bnode = readMLP(256, 256);
  MLP2 p_edges = readMLP(11, 32);
  MLP2 p_bedge[4], p_bnode[4];
  for (int i = 0; i < 4; ++i) { p_bedge[i] = readMLP(384, 384); p_bnode[i] = readMLP(256, 256); }
  MLP2 b_edges = readMLP(11, 32);
  MLP2 b_bedge = readMLP(384, 384);
  MLP2 b_bnode = readMLP(256, 256);
  // extract (no LayerNorm): W0, W1, b0, b1
  const float* exW0f = F(ix++); const float* exW1f = F(ix++);
  const float* exb0  = F(ix++); const float* exb1f = F(ix++);
  const bf16* exW0 = cvtW(exW0f, 128, 128, 128, 128);
  const bf16* exW1 = cvtW(exW1f, 128, 72, 128, 80);
  float* exb1 = (float*)alloc(80 * sizeof(float));
  pad_b_kernel<<<1, 128, 0, stream>>>(exb1f, 72, exb1, 80);

  // ---- activation buffers ----
  bf16*  x_era_in = (bf16*) alloc((size_t)cN_ERA * 192 * 2);
  bf16*  x_h_in   = (bf16*) alloc((size_t)cN_H   * 32  * 2);
  bf16*  edge_in  = (bf16*) alloc((size_t)cE_E2H * 32  * 2);
  float* x_era    = (float*)alloc((size_t)cN_ERA * 128 * 4);
  float* x_h      = (float*)alloc((size_t)cN_H   * 128 * 4);
  float* x_proc   = (float*)alloc((size_t)cN_H   * 128 * 4);
  float* eaA      = (float*)alloc((size_t)cE_E2H * 128 * 4);
  float* eaB      = (float*)alloc((size_t)cE_E2H * 128 * 4);
  bf16*  hiddenE  = (bf16*) alloc((size_t)cE_E2H * 128 * 2);
  bf16*  hiddenN  = (bf16*) alloc((size_t)cN_ERA * 128 * 2);
  float* agg      = (float*)alloc((size_t)cN_ERA * 128 * 4);

  float* ea_cur = eaA;
  float* ea_alt = eaB;

  // ---- launch helpers ----
  auto shGemm = [](int Kpad) -> size_t {
    size_t m = (size_t)MT * (Kpad + 8) * 2;
    size_t e = (size_t)MT * EPITCH * 4;
    return m > e ? m : e;
  };
  auto gemm = [&](const void* A, int a_f32, int lda, const bf16* W,
                  int Kpad, int Npad, int N, const float* bias, int act,
                  const float* g, const float* bln,
                  const float* res, int res_ld,
                  float* oF, int ldoF, bf16* oH, int ldoH, int M) {
    dim3 grid((M + MT - 1) / MT);
    size_t sh = shGemm(Kpad);
    if (Npad == 128)
      gemm_kernel<8><<<grid, TPB, sh, stream>>>(A, a_f32, lda, W, Kpad, Npad, N,
          bias, act, g, bln, res, res_ld, oF, ldoF, oH, ldoH, M);
    else
      gemm_kernel<5><<<grid, TPB, sh, stream>>>(A, a_f32, lda, W, Kpad, Npad, N,
          bias, act, g, bln, res, res_ld, oF, ldoF, oH, ldoH, M);
  };
  auto zero = [&](float* p, int n) {
    zero_f32_kernel<<<(n + 255) / 256, 256, 0, stream>>>(p, n);
  };
  const size_t shEdge = shGemm(384);   // 50176 B
  const size_t shNode = shGemm(256);   // 33792 B

  // one full message-passing block
  auto mp_block = [&](const float* xsrc, float* xdst, const int* ei, int E,
                      int n_dst, const MLP2& be, const MLP2& bn) {
    // edge MLP
    edge_l1_kernel<<<dim3((E + MT - 1) / MT), TPB, shEdge, stream>>>(
        ei, E, xsrc, xdst, ea_cur, be.W0, be.b0, hiddenE);
    gemm(hiddenE, 0, 128, be.W1, 128, 128, 128, be.b1, 1, be.g, be.bln,
         ea_cur, 128, ea_alt, 128, nullptr, 0, E);
    { float* t = ea_cur; ea_cur = ea_alt; ea_alt = t; }
    // aggregate
    zero(agg, n_dst * 128);
    scatter_add_kernel<<<(unsigned)(((size_t)E * 128 + 255) / 256), 256, 0, stream>>>(
        ea_cur, ei, E, agg);
    // node MLP (in-place residual update of xdst)
    node_l1_kernel<<<dim3((n_dst + MT - 1) / MT), TPB, shNode, stream>>>(
        xdst, agg, bn.W0, bn.b0, hiddenN, n_dst);
    gemm(hiddenN, 0, 128, bn.W1, 128, 128, 128, bn.b1, 1, bn.g, bn.bln,
         xdst, 128, xdst, 128, nullptr, 0, n_dst);
  };

  // ======================= pipeline ========================================
  // input prep
  prep_era_in_kernel<<<(cN_ERA * 192 + 255) / 256, 256, 0, stream>>>(
      x, era_ll, era_tr, x_era_in);
  prep_h_in_kernel<<<(cN_H * 32 + 255) / 256, 256, 0, stream>>>(
      h_ll, h_tr, x_h_in);

  // ---- forward mapper (era -> h) ----
  gemm(x_era_in, 0, 192, f_src.W0, 192, 128, 128, f_src.b0, 1, nullptr, nullptr,
       nullptr, 0, nullptr, 0, hiddenN, 128, cN_ERA);
  gemm(hiddenN, 0, 128, f_src.W1, 128, 128, 128, f_src.b1, 1, f_src.g, f_src.bln,
       nullptr, 0, x_era, 128, nullptr, 0, cN_ERA);
  gemm(x_h_in, 0, 32, f_dst.W0, 32, 128, 128, f_dst.b0, 1, nullptr, nullptr,
       nullptr, 0, nullptr, 0, hiddenN, 128, cN_H);
  gemm(hiddenN, 0, 128, f_dst.W1, 128, 128, 128, f_dst.b1, 1, f_dst.g, f_dst.bln,
       nullptr, 0, x_h, 128, nullptr, 0, cN_H);
  prep_edge_in_kernel<<<(cE_E2H * 32 + 255) / 256, 256, 0, stream>>>(
      e2h_at, e2h_tr, edge_in, cE_E2H);
  gemm(edge_in, 0, 32, f_edges.W0, 32, 128, 128, f_edges.b0, 1, nullptr, nullptr,
       nullptr, 0, nullptr, 0, hiddenE, 128, cE_E2H);
  gemm(hiddenE, 0, 128, f_edges.W1, 128, 128, 128, f_edges.b1, 1, f_edges.g, f_edges.bln,
       nullptr, 0, ea_cur, 128, nullptr, 0, cE_E2H);
  mp_block(x_era, x_h, e2h_ei, cE_E2H, cN_H, f_bedge, f_bnode);

  // ---- processor (h -> h, 4 blocks, residual around the stack) ----
  copy_f32_kernel<<<(cN_H * 128 + 255) / 256, 256, 0, stream>>>(
      x_proc, x_h, cN_H * 128);
  prep_edge_in_kernel<<<(cE_H2H * 32 + 255) / 256, 256, 0, stream>>>(
      h2h_at, h2h_tr, edge_in, cE_H2H);
  gemm(edge_in, 0, 32, p_edges.W0, 32, 128, 128, p_edges.b0, 1, nullptr, nullptr,
       nullptr, 0, nullptr, 0, hiddenE, 128, cE_H2H);
  gemm(hiddenE, 0, 128, p_edges.W1, 128, 128, 128, p_edges.b1, 1, p_edges.g, p_edges.bln,
       nullptr, 0, ea_cur, 128, nullptr, 0, cE_H2H);
  for (int i = 0; i < 4; ++i)
    mp_block(x_proc, x_proc, h2h_ei, cE_H2H, cN_H, p_bedge[i], p_bnode[i]);
  addto_f32_kernel<<<(cN_H * 128 + 255) / 256, 256, 0, stream>>>(
      x_proc, x_h, cN_H * 128);

  // ---- backward mapper (h -> era) ----
  prep_edge_in_kernel<<<(cE_H2E * 32 + 255) / 256, 256, 0, stream>>>(
      h2e_at, h2e_tr, edge_in, cE_H2E);
  gemm(edge_in, 0, 32, b_edges.W0, 32, 128, 128, b_edges.b0, 1, nullptr, nullptr,
       nullptr, 0, nullptr, 0, hiddenE, 128, cE_H2E);
  gemm(hiddenE, 0, 128, b_edges.W1, 128, 128, 128, b_edges.b1, 1, b_edges.g, b_edges.bln,
       nullptr, 0, ea_cur, 128, nullptr, 0, cE_H2E);
  mp_block(x_proc, x_era, h2e_ei, cE_H2E, cN_ERA, b_bedge, b_bnode);

  // ---- extract + final residual (out = extract(x_era) + x[:, -1, :, :72]) ----
  gemm(x_era, 1, 128, exW0, 128, 128, 128, exb0, 1, nullptr, nullptr,
       nullptr, 0, nullptr, 0, hiddenN, 128, cN_ERA);
  gemm(hiddenN, 0, 128, exW1, 128, 80, 72, exb1, 0, nullptr, nullptr,
       x + (size_t)cN_ERA * 80, 80, (float*)d_out, 72, nullptr, 0, cN_ERA);
}
